// GATClassifier_13597866459806
// MI455X (gfx1250) — compile-verified, matrix operand order checked by reading, not run on hardware
//
#include <hip/hip_runtime.h>
#include <hip/hip_bf16.h>
#include <cstdint>

// ---------------- WMMA types ----------------
typedef __attribute__((ext_vector_type(16))) _Float16 v16h;
typedef __attribute__((ext_vector_type(8)))  float    v8f;

#define LEAKY_SLOPE 0.2f

// =====================================================================
// WMMA GEMM:  C[M,NOUT] = A[M,K] * B[K,NOUT]   (fp32 in, f16 WMMA, fp32 out)
// Block: 32*NT threads (NT = NOUT/16 waves). Wave w owns column tile w and
// sweeps MWAVE consecutive 16-row M-tiles.
// B is staged once per block into LDS *pre-swizzled into WMMA fragment
// order* ([ntile][ktile][lane][16 halves]) so each fragment is a single
// contiguous 32B v16h read (2x ds_load_b128) issued right before its WMMA
// (short live ranges -> no scratch spills). A is read as float4 pairs and
// converted with v_cvt_pk_f16_f32. M-tail guard is wave-uniform -> EXEC
// stays all-ones at every WMMA.
// =====================================================================
template<int K, int NOUT, int MWAVE>
__global__ void __launch_bounds__(32 * (NOUT / 16), 1)
gemm_wmma(const float* __restrict__ A,
          const float* __restrict__ B,
          float* __restrict__ C, int M) {
  constexpr int NT = NOUT / 16;
  constexpr int KT = K / 32;
  __shared__ _Float16 Bs[NT * KT * 32 * 16];   // == K*NOUT halves

  const int tid = threadIdx.x;
  // Stage + swizzle B: dest idx = ((nt*KT + kt)*32 + lane)*16 + i
  for (int idx = tid; idx < NT * KT * 512; idx += 32 * NT) {
    const int i  = idx & 15;
    const int l  = (idx >> 4) & 31;
    const int g  = idx >> 9;           // nt*KT + kt
    const int kt = g % KT;
    const int nt = g / KT;
    const int col = nt * 16 + (l & 15);
    const int kbs = (l >> 4) * 8;
    const int k   = kt * 32 + ((i < 8) ? (kbs + i) : (16 + kbs + i - 8));
    Bs[idx] = (_Float16)B[k * NOUT + col];
  }
  __syncthreads();

  const int wave = tid >> 5;
  const int lane = tid & 31;
  const int l15  = lane & 15;          // A: row in tile; B/C/D: col in tile
  const int kb   = (lane >> 4) * 8;    // K-half select (ISA 16-bit A layout)
  const int ro   = (lane >> 4) * 8;    // C/D row offset for hi lanes
  const int nbase = wave * 16;
  // Base of this wave's fragment region in LDS (contiguous per ktile).
  const _Float16* __restrict__ bsw = Bs + ((size_t)(wave * KT) * 32 + lane) * 16;

#pragma unroll
  for (int mi = 0; mi < MWAVE; ++mi) {
    const int mbase = (blockIdx.x * MWAVE + mi) * 16;
    if (mbase >= M) break;             // uniform across wave
    const float4* __restrict__ A4 = (const float4*)(A + (size_t)(mbase + l15) * K);
    v8f acc = {};
#pragma unroll
    for (int kt = 0; kt < KT; ++kt) {
      const int o = (kt * 32 + kb) >> 2;           // kb in {0,8}: exact
      const float4 a0 = A4[o], a1 = A4[o + 1];     // K = kt*32 + kb .. +7
      const float4 a2 = A4[o + 4], a3 = A4[o + 5]; // K = kt*32 + 16 + kb .. +7
      v16h a;
      a[0]=(_Float16)a0.x; a[1]=(_Float16)a0.y; a[2]=(_Float16)a0.z; a[3]=(_Float16)a0.w;
      a[4]=(_Float16)a1.x; a[5]=(_Float16)a1.y; a[6]=(_Float16)a1.z; a[7]=(_Float16)a1.w;
      a[8]=(_Float16)a2.x; a[9]=(_Float16)a2.y; a[10]=(_Float16)a2.z; a[11]=(_Float16)a2.w;
      a[12]=(_Float16)a3.x; a[13]=(_Float16)a3.y; a[14]=(_Float16)a3.z; a[15]=(_Float16)a3.w;
      const v16h b = *(const v16h*)(bsw + (size_t)kt * 32 * 16);
      acc = __builtin_amdgcn_wmma_f32_16x16x32_f16(
          /*neg_a=*/false, a, /*neg_b=*/false, b,
          /*c_mod=*/(short)0, acc, /*reuse_a=*/false, /*reuse_b=*/false);
    }
    float* __restrict__ Crow = C + (size_t)(mbase + ro) * NOUT + nbase + l15;
#pragma unroll
    for (int v = 0; v < 8; ++v) Crow[(size_t)v * NOUT] = acc[v];
  }
}

// =====================================================================
// a_s[n,h] = sum_c h[n,h*C+c]*att_src[h,c] ; same for a_d
// =====================================================================
__global__ void attn_coeff(const float* __restrict__ h,
                           const float* __restrict__ att_s,
                           const float* __restrict__ att_d,
                           float* __restrict__ a_s, float* __restrict__ a_d,
                           int N, int H, int C) {
  int idx = blockIdx.x * blockDim.x + threadIdx.x;
  if (idx >= N * H) return;
  int n = idx / H, hh = idx % H;
  const float* __restrict__ row = h + (size_t)n * H * C + hh * C;
  float s = 0.f, d = 0.f;
  for (int c = 0; c < C; ++c) {
    float v = row[c];
    s += v * att_s[hh * C + c];
    d += v * att_d[hh * C + c];
  }
  a_s[idx] = s;
  a_d[idx] = d;
}

__global__ void fill_f32(float* __restrict__ p, float v, size_t n) {
  size_t i = (size_t)blockIdx.x * blockDim.x + threadIdx.x;
  if (i < n) p[i] = v;
}

// float atomic max via sign-split int trick (correct with init = -inf)
__device__ __forceinline__ void atomicMaxFloat(float* addr, float val) {
  if (val >= 0.f) atomicMax((int*)addr, __float_as_int(val));
  else            atomicMin((unsigned int*)addr, __float_as_uint(val));
}

// =====================================================================
// Pass A: e = leaky_relu(a_s[src]+a_d[dst]); store e; segment max into m[dst]
// Edges [0,E) from arrays, [E,E+N) are self-loops.
// =====================================================================
__global__ void edge_max(const int* __restrict__ src, const int* __restrict__ dst,
                         int E, int N, int H,
                         const float* __restrict__ a_s, const float* __restrict__ a_d,
                         float* __restrict__ m, float* __restrict__ estore) {
  int e = blockIdx.x * blockDim.x + threadIdx.x;
  int Ep = E + N;
  if (e >= Ep) return;
  int s = (e < E) ? src[e] : (e - E);
  int d = (e < E) ? dst[e] : (e - E);
  for (int h = 0; h < H; ++h) {
    float v = a_s[s * H + h] + a_d[d * H + h];
    v = (v > 0.f) ? v : LEAKY_SLOPE * v;
    estore[(size_t)e * H + h] = v;
    atomicMaxFloat(&m[d * H + h], v);
  }
}

// =====================================================================
// Pass B: ex = exp(e - m[dst]); store ex; segment sum into dn[dst]
// =====================================================================
__global__ void edge_expsum(const int* __restrict__ dst, int E, int N, int H,
                            const float* __restrict__ m,
                            float* __restrict__ estore, float* __restrict__ dn) {
  int e = blockIdx.x * blockDim.x + threadIdx.x;
  int Ep = E + N;
  if (e >= Ep) return;
  int d = (e < E) ? dst[e] : (e - E);
  for (int h = 0; h < H; ++h) {
    float ex = __expf(estore[(size_t)e * H + h] - m[d * H + h]);
    estore[(size_t)e * H + h] = ex;
    atomicAdd(&dn[d * H + h], ex);
  }
}

// =====================================================================
// Pass C: one wave per edge. out[dst,c] += alpha(head(c)) * h[src,c]
// F = total feature width (H*C). Each lane owns F/32 contiguous channels
// (all within one head since 64 % (F/32) == 0). Gather loads hoisted so
// they coalesce to b128/b64 before the f32 atomics.
// =====================================================================
template<int F, int H>
__global__ void edge_aggregate(const int* __restrict__ src, const int* __restrict__ dst,
                               int E, int N,
                               const float* __restrict__ h,
                               const float* __restrict__ estore,
                               const float* __restrict__ dn,
                               float* __restrict__ out) {
  int gid  = blockIdx.x * blockDim.x + threadIdx.x;
  int wid  = gid >> 5;
  int lane = threadIdx.x & 31;
  int Ep = E + N;
  if (wid >= Ep) return;
  int s = (wid < E) ? src[wid] : (wid - E);
  int d = (wid < E) ? dst[wid] : (wid - E);
  constexpr int CPL = F / 32;
  const int head = (lane * CPL) / (F / H);
  const float alpha = estore[(size_t)wid * H + head] / dn[d * H + head];
  const float* __restrict__ hs = h   + (size_t)s * F + lane * CPL;
  float* __restrict__        od = out + (size_t)d * F + lane * CPL;
  float tmp[CPL];
#pragma unroll
  for (int i = 0; i < CPL; ++i) tmp[i] = hs[i];
#pragma unroll
  for (int i = 0; i < CPL; ++i) atomicAdd(&od[i], alpha * tmp[i]);
}

__global__ void bias_elu(float* __restrict__ x, const float* __restrict__ b,
                         size_t total, int F) {
  size_t i = (size_t)blockIdx.x * blockDim.x + threadIdx.x;
  if (i >= total) return;
  float v = x[i] + b[i % F];
  x[i] = (v > 0.f) ? v : (__expf(v) - 1.f);
}

// out[n,c] = h[n,:64] @ Wl[:,c] + bl[c]   (CLS = 10)
__global__ void linear_head(const float* __restrict__ h, const float* __restrict__ Wl,
                            const float* __restrict__ bl, float* __restrict__ out, int N) {
  int idx = blockIdx.x * blockDim.x + threadIdx.x;
  if (idx >= N * 10) return;
  int n = idx / 10, c = idx % 10;
  const float* __restrict__ row = h + (size_t)n * 64;
  float acc = bl[c];
#pragma unroll 8
  for (int k = 0; k < 64; ++k) acc += row[k] * Wl[k * 10 + c];
  out[idx] = acc;
}

static inline unsigned cdiv(size_t a, unsigned b) { return (unsigned)((a + b - 1) / b); }

extern "C" void kernel_launch(void* const* d_in, const int* in_sizes, int n_in,
                              void* d_out, int out_size, void* d_ws, size_t ws_size,
                              hipStream_t stream) {
  // inputs (setup_inputs order)
  const float* x   = (const float*)d_in[0];
  const int*   ei  = (const int*)  d_in[1];   // [2,E] flat: row0 = src, row1 = dst
  const float* W1  = (const float*)d_in[2];
  const float* as1 = (const float*)d_in[3];
  const float* ad1 = (const float*)d_in[4];
  const float* b1  = (const float*)d_in[5];
  const float* W2  = (const float*)d_in[6];
  const float* as2 = (const float*)d_in[7];
  const float* ad2 = (const float*)d_in[8];
  const float* b2  = (const float*)d_in[9];
  const float* Wl  = (const float*)d_in[10];
  const float* bl  = (const float*)d_in[11];
  float* out = (float*)d_out;

  const int N  = in_sizes[0] / 256;   // IN = 256
  const int E  = in_sizes[1] / 2;
  const int Ep = E + N;
  const int* esrc = ei;
  const int* edst = ei + E;
  const int mtiles = (N + 15) / 16;

  // workspace layout (floats), buffers reused across layers
  float* ws = (float*)d_ws;
  size_t off = 0;
  float* bufA = ws + off; off += (size_t)N * 128;  // h1; later h2 (N*64) + out2 (N*64)
  float* bufB = ws + off; off += (size_t)N * 128;  // out1 / layer-2 input
  float* aS   = ws + off; off += (size_t)N * 2;
  float* aD   = ws + off; off += (size_t)N * 2;
  float* mM   = ws + off; off += (size_t)N * 2;
  float* dN   = ws + off; off += (size_t)N * 2;
  float* eSt  = ws + off; off += (size_t)Ep * 2;   // per-edge e -> exp(e-m)
  (void)ws_size; (void)n_in; (void)out_size;

  const unsigned TB = 256;

  // ---------------- Layer 1 (H=2, C=64, F=128) ----------------
  gemm_wmma<256, 128, 4><<<cdiv(mtiles, 4), 256, 0, stream>>>(x, W1, bufA, N);
  attn_coeff<<<cdiv((size_t)N * 2, TB), TB, 0, stream>>>(bufA, as1, ad1, aS, aD, N, 2, 64);
  fill_f32<<<cdiv((size_t)N * 2, TB), TB, 0, stream>>>(mM, -INFINITY, (size_t)N * 2);
  fill_f32<<<cdiv((size_t)N * 2, TB), TB, 0, stream>>>(dN, 0.f, (size_t)N * 2);
  fill_f32<<<cdiv((size_t)N * 128, TB), TB, 0, stream>>>(bufB, 0.f, (size_t)N * 128);
  edge_max<<<cdiv(Ep, TB), TB, 0, stream>>>(esrc, edst, E, N, 2, aS, aD, mM, eSt);
  edge_expsum<<<cdiv(Ep, TB), TB, 0, stream>>>(edst, E, N, 2, mM, eSt, dN);
  edge_aggregate<128, 2><<<cdiv((size_t)Ep * 32, TB), TB, 0, stream>>>(
      esrc, edst, E, N, bufA, eSt, dN, bufB);
  bias_elu<<<cdiv((size_t)N * 128, TB), TB, 0, stream>>>(bufB, b1, (size_t)N * 128, 128);

  // ---------------- Layer 2 (H=1, C=64, F=64) ----------------
  float* h2   = bufA;                  // N*64
  float* out2 = bufA + (size_t)N * 64; // N*64
  gemm_wmma<128, 64, 4><<<cdiv(mtiles, 4), 128, 0, stream>>>(bufB, W2, h2, N);
  attn_coeff<<<cdiv((size_t)N, TB), TB, 0, stream>>>(h2, as2, ad2, aS, aD, N, 1, 64);
  fill_f32<<<cdiv((size_t)N, TB), TB, 0, stream>>>(mM, -INFINITY, (size_t)N);
  fill_f32<<<cdiv((size_t)N, TB), TB, 0, stream>>>(dN, 0.f, (size_t)N);
  fill_f32<<<cdiv((size_t)N * 64, TB), TB, 0, stream>>>(out2, 0.f, (size_t)N * 64);
  edge_max<<<cdiv(Ep, TB), TB, 0, stream>>>(esrc, edst, E, N, 1, aS, aD, mM, eSt);
  edge_expsum<<<cdiv(Ep, TB), TB, 0, stream>>>(edst, E, N, 1, mM, eSt, dN);
  edge_aggregate<64, 1><<<cdiv((size_t)Ep * 32, TB), TB, 0, stream>>>(
      esrc, edst, E, N, h2, eSt, dN, out2);
  bias_elu<<<cdiv((size_t)N * 64, TB), TB, 0, stream>>>(out2, b2, (size_t)N * 64, 64);

  // ---------------- Linear head ----------------
  linear_head<<<cdiv((size_t)N * 10, TB), TB, 0, stream>>>(out2, Wl, bl, out, N);
}
